// PixelContrastiveLearning_25297357373846
// MI455X (gfx1250) — matrix operand order repaired and checked by reading, not compiled, not run on hardware
//
#include <hip/hip_runtime.h>
#include <hip/hip_bf16.h>
#include <math.h>

// ---------------------------------------------------------------------------
// Problem constants (match reference)
// ---------------------------------------------------------------------------
#define BATCH 16
#define CIN 2048
#define HW 4096            // 64*64
#define NPIX 65536         // BATCH*HW
#define PD 128             // projection dim
#define AD 64              // attention hidden dim
#define NS 64              // MAX_SAMPLES
#define TEMP 0.1f
#define BASE_TEMP 0.07f
#define SIM_TH 0.7f
#define LOSS_W 0.1f
#define BN_EPS 1e-5f

typedef __attribute__((ext_vector_type(16))) __bf16 v16bf;
typedef __attribute__((ext_vector_type(8)))  float  v8f;
typedef __attribute__((ext_vector_type(4)))  int    v4i;

typedef __attribute__((address_space(1))) v4i*  gv4i_p;
typedef __attribute__((address_space(3))) v4i*  lv4i_p;
typedef __attribute__((address_space(1))) void* gvoid_p;
typedef __attribute__((address_space(3))) void* lvoid_p;

// Async global->LDS path availability (guarded; falls back to direct loads).
#if __has_builtin(__builtin_amdgcn_global_load_async_to_lds_b128)
#define K1_ASYNC 1
#else
#define K1_ASYNC 0
#endif

template <int N>
__device__ __forceinline__ void wait_asynccnt() {
#if __has_builtin(__builtin_amdgcn_s_wait_asynccnt)
    __builtin_amdgcn_s_wait_asynccnt(N);
#else
    asm volatile("s_wait_asynccnt %0" ::"i"(N) : "memory");
#endif
}

__device__ __forceinline__ void async_g2l_b128(const float* g, float* l) {
#if K1_ASYNC
    __builtin_amdgcn_global_load_async_to_lds_b128(
        (gv4i_p)(gvoid_p)g, (lv4i_p)(lvoid_p)l, 0, 0);
#endif
}

// A-matrix (16x32 bf16) element K index for vgpr-slot j (j = 2*v+e) and lane half hl.
__device__ __forceinline__ int a_kidx(int j, int hl) {
    int v = j >> 1, e = j & 1;
    return hl * 16 + 2 * v + e;
}
// B-matrix (32x16 bf16) element K index for slot j and lane half hl.
__device__ __forceinline__ int b_kidx(int j, int hl) {
    int v = j >> 1, e = j & 1;
    return hl * 16 + 2 * v + e;
}

// ---------------------------------------------------------------------------
// Workspace layout (bytes)
// ---------------------------------------------------------------------------
#define OFF_WSWZ   ((size_t)0)          // 12*64*512 bf16  = 786432 B  (W1 | Wa1 swizzled)
#define OFF_W2SWZ  ((size_t)786432)     // 8*4*512 bf16    =  32768 B
#define OFF_SCALE  ((size_t)819200)     // 128 f32
#define OFF_SHIFT  ((size_t)819712)     // 128 f32
#define OFF_ATTN   ((size_t)820224)     // 65536 f32       = 262144 B
#define OFF_H      ((size_t)1082368)    // 128*65536 f32   = 33554432 B
#define OFF_F      ((size_t)34636800)   // 65536*128 f32   = 33554432 B
// total ~68.2 MB

// ---------------------------------------------------------------------------
// Kernel 0: swizzle weights fp32 -> bf16 in WMMA A-fragment lane order.
// Fragment (tile, kblock): 32 lanes x 16 bf16 contiguous (1 KB).
//   tiles 0..7  : W1  rows 16t..16t+15   (K = 2048, 64 kblocks)
//   tiles 8..11 : Wa1 rows 16(t-8)..     (K = 2048, 64 kblocks)
//   w2 tiles 0..7 : W2 rows              (K = 128, 4 kblocks)
// ---------------------------------------------------------------------------
__global__ __launch_bounds__(32) void swizzle_weights(
    const float* __restrict__ W1, const float* __restrict__ Wa1,
    const float* __restrict__ W2, __bf16* __restrict__ wswz,
    __bf16* __restrict__ w2swz) {
    int blk = blockIdx.x, lane = threadIdx.x;
    int hl = lane >> 4, m = lane & 15;
    if (blk < 12 * 64) {
        int t = blk / 64, kb = blk % 64;
        const float* src = (t < 8) ? (W1 + (size_t)(16 * t + m) * CIN)
                                   : (Wa1 + (size_t)(16 * (t - 8) + m) * CIN);
        __bf16* dst = wswz + ((size_t)blk * 32 + lane) * 16;
#pragma unroll
        for (int j = 0; j < 16; ++j) {
            int k = kb * 32 + a_kidx(j, hl);
            dst[j] = (__bf16)src[k];
        }
    } else {
        int bb = blk - 12 * 64;            // 0..31
        int t = bb / 4, kb = bb % 4;
        const float* src = W2 + (size_t)(16 * t + m) * PD;
        __bf16* dst = w2swz + ((size_t)bb * 32 + lane) * 16;
#pragma unroll
        for (int j = 0; j < 16; ++j) {
            int k = kb * 32 + a_kidx(j, hl);
            dst[j] = (__bf16)src[k];
        }
    }
}

// ---------------------------------------------------------------------------
// Kernel 1: fused conv1 (2048->128) + attention head (2048->64 ->relu-> dot
// with Wa2 -> sigmoid).  One wave per 16-pixel tile; 12 WMMA accumulators;
// K-loop = 64 x bf16 16x16x32 WMMA per tile.
// feats slabs (32 ch x 16 px fp32 = 2 KB) are double-buffered in LDS via
// ASYNCcnt-tracked global_load_async_to_lds_b128 when available.
// Writes: h (pre-BN, +b1) as [128][65536] f32, attn weights [65536] f32.
// ---------------------------------------------------------------------------
__global__ __launch_bounds__(32) void conv1_attn(
    const float* __restrict__ feats, const float* __restrict__ b1,
    const float* __restrict__ ba1, const float* __restrict__ Wa2,
    const float* __restrict__ ba2, const __bf16* __restrict__ wswz,
    float* __restrict__ hbuf, float* __restrict__ attnw) {
    int blk = blockIdx.x;              // 0..4095 : 16-pixel tile
    int lane = threadIdx.x;
    int hl = lane >> 4, col = lane & 15;
    int img = blk >> 8;                // 256 tiles per image
    int hw0 = (blk & 255) * 16;
    const float* fb0 = feats + (size_t)img * CIN * HW + hw0;     // tile base
    const float* fb = fb0 + col;                                  // lane column

    v8f acc[12];
#pragma unroll
    for (int t = 0; t < 12; ++t) acc[t] = (v8f){0.f, 0.f, 0.f, 0.f, 0.f, 0.f, 0.f, 0.f};

#if K1_ASYNC
    // stage[buf] : 32 rows (channels) x 16 px fp32 = 2 KB
    __shared__ float stage[2][32 * 16];
    // Each lane moves 4 x 16B segments per slab: transfer id = t*32 + lane,
    // row = id/4, seg = id%4 (global rows are 16 KB apart, segs contiguous).
    {
#pragma unroll
        for (int t = 0; t < 4; ++t) {
            int id = t * 32 + lane;
            int row = id >> 2, seg = id & 3;
            async_g2l_b128(fb0 + ((size_t)row * HW) + seg * 4,
                           &stage[0][row * 16 + seg * 4]);
        }
    }
    for (int kb = 0; kb < 64; ++kb) {
        int buf = kb & 1;
        if (kb + 1 < 64) {
#pragma unroll
            for (int t = 0; t < 4; ++t) {
                int id = t * 32 + lane;
                int row = id >> 2, seg = id & 3;
                async_g2l_b128(fb0 + ((size_t)((kb + 1) * 32 + row) * HW) + seg * 4,
                               &stage[buf ^ 1][row * 16 + seg * 4]);
            }
            wait_asynccnt<4>();        // slab kb complete, kb+1 in flight
        } else {
            wait_asynccnt<0>();
        }
        // build B fragment (32 channels x 16 pixels) from LDS, convert bf16
        v16bf bfrag;
#pragma unroll
        for (int v = 0; v < 8; ++v) {
            int k0 = hl * 16 + 2 * v;
            float x0 = stage[buf][k0 * 16 + col];
            float x1 = stage[buf][(k0 + 1) * 16 + col];
            bfrag[2 * v] = (__bf16)x0;
            bfrag[2 * v + 1] = (__bf16)x1;
        }
#pragma unroll
        for (int t = 0; t < 12; ++t) {
            v16bf af = *(const v16bf*)(wswz + ((size_t)(t * 64 + kb) * 32 + lane) * 16);
            acc[t] = __builtin_amdgcn_wmma_f32_16x16x32_bf16(
                false, af, false, bfrag, (short)0, acc[t], false, false);
        }
    }
#else
    for (int kb = 0; kb < 64; ++kb) {
        // prefetch the next K-slab of feats
        if (kb + 1 < 64)
            __builtin_prefetch(fb + (size_t)((kb + 1) * 32 + hl * 16) * HW, 0, 1);
        // build B fragment (32 channels x 16 pixels) in bf16
        v16bf bfrag;
#pragma unroll
        for (int v = 0; v < 8; ++v) {
            int k0 = kb * 32 + hl * 16 + 2 * v;
            float x0 = fb[(size_t)k0 * HW];
            float x1 = fb[(size_t)(k0 + 1) * HW];
            bfrag[2 * v] = (__bf16)x0;
            bfrag[2 * v + 1] = (__bf16)x1;
        }
#pragma unroll
        for (int t = 0; t < 12; ++t) {
            v16bf af = *(const v16bf*)(wswz + ((size_t)(t * 64 + kb) * 32 + lane) * 16);
            acc[t] = __builtin_amdgcn_wmma_f32_16x16x32_bf16(
                false, af, false, bfrag, (short)0, acc[t], false, false);
        }
    }
#endif

    int p = img * HW + hw0 + col;      // this lane's pixel
    // projection pre-BN (+b1) -> hbuf[ch][pixel]
#pragma unroll
    for (int t = 0; t < 8; ++t) {
#pragma unroll
        for (int r = 0; r < 8; ++r) {
            int ch = 16 * t + r + 8 * hl;
            hbuf[(size_t)ch * NPIX + p] = acc[t][r] + b1[ch];
        }
    }
    // attention: relu(a + ba1) dot Wa2, cross-lane combine, sigmoid
    float s = 0.f;
#pragma unroll
    for (int t = 8; t < 12; ++t) {
#pragma unroll
        for (int r = 0; r < 8; ++r) {
            int ca = 16 * (t - 8) + r + 8 * hl;
            float v = acc[t][r] + ba1[ca];
            v = fmaxf(v, 0.f);
            s += Wa2[ca] * v;
        }
    }
    s += __shfl_xor(s, 16, 32);
    float sig = 1.f / (1.f + __expf(-(s + ba2[0])));
    if (hl == 0) attnw[p] = sig;
}

// ---------------------------------------------------------------------------
// Kernel 2: BN statistics per channel over all 65536 pixels -> fused
// scale = gamma*rsqrt(var+eps), shift = beta - mean*scale.
// ---------------------------------------------------------------------------
__global__ __launch_bounds__(256) void bn_stats(
    const float* __restrict__ hbuf, const float* __restrict__ gamma,
    const float* __restrict__ beta, float* __restrict__ scale,
    float* __restrict__ shift) {
    int ch = blockIdx.x, tid = threadIdx.x;
    const float* src = hbuf + (size_t)ch * NPIX;
    float s = 0.f, sq = 0.f;
    for (int i = tid; i < NPIX; i += 256) {
        float v = src[i];
        s += v;
        sq += v * v;
    }
    __shared__ float ss[256], sqq[256];
    ss[tid] = s; sqq[tid] = sq;
    __syncthreads();
    for (int off = 128; off > 0; off >>= 1) {
        if (tid < off) { ss[tid] += ss[tid + off]; sqq[tid] += sqq[tid + off]; }
        __syncthreads();
    }
    if (tid == 0) {
        float mean = ss[0] / (float)NPIX;
        float var = sqq[0] / (float)NPIX - mean * mean;
        float r = rsqrtf(var + BN_EPS);
        float sc = gamma[ch] * r;
        scale[ch] = sc;
        shift[ch] = beta[ch] - mean * sc;
    }
}

// ---------------------------------------------------------------------------
// Kernel 3: conv2 (128->128) on relu(BN(h)) with WMMA, +b2, per-pixel L2
// normalize.  Writes f as [pixel][128] f32 (row per pixel = 512 B).
// ---------------------------------------------------------------------------
__global__ __launch_bounds__(32) void conv2_norm(
    const float* __restrict__ hbuf, const float* __restrict__ scale,
    const float* __restrict__ shift, const float* __restrict__ b2,
    const __bf16* __restrict__ w2swz, float* __restrict__ fbuf) {
    int blk = blockIdx.x;              // 0..4095
    int lane = threadIdx.x;
    int hl = lane >> 4, col = lane & 15;
    int p = blk * 16 + col;            // absolute pixel

    v8f acc[8];
#pragma unroll
    for (int t = 0; t < 8; ++t) acc[t] = (v8f){0.f, 0.f, 0.f, 0.f, 0.f, 0.f, 0.f, 0.f};

#pragma unroll
    for (int kb = 0; kb < 4; ++kb) {
        v16bf bfrag;
#pragma unroll
        for (int v = 0; v < 8; ++v) {
#pragma unroll
            for (int e = 0; e < 2; ++e) {
                int ch = kb * 32 + hl * 16 + 2 * v + e;
                float x = hbuf[(size_t)ch * NPIX + p] * scale[ch] + shift[ch];
                bfrag[2 * v + e] = (__bf16)fmaxf(x, 0.f);
            }
        }
#pragma unroll
        for (int t = 0; t < 8; ++t) {
            v16bf af = *(const v16bf*)(w2swz + ((size_t)(t * 4 + kb) * 32 + lane) * 16);
            acc[t] = __builtin_amdgcn_wmma_f32_16x16x32_bf16(
                false, af, false, bfrag, (short)0, acc[t], false, false);
        }
    }

    float vals[64];
    float ssq = 0.f;
#pragma unroll
    for (int t = 0; t < 8; ++t) {
#pragma unroll
        for (int r = 0; r < 8; ++r) {
            int ch = 16 * t + r + 8 * hl;
            float v = acc[t][r] + b2[ch];
            vals[t * 8 + r] = v;
            ssq += v * v;
        }
    }
    ssq += __shfl_xor(ssq, 16, 32);
    float rn = 1.f / fmaxf(sqrtf(ssq), 1e-12f);
#pragma unroll
    for (int t = 0; t < 8; ++t) {
#pragma unroll
        for (int r = 0; r < 8; ++r) {
            int ch = 16 * t + r + 8 * hl;
            fbuf[(size_t)p * PD + ch] = vals[t * 8 + r] * rn;
        }
    }
}

// ---------------------------------------------------------------------------
// Kernel 4: per-image contrastive loss.  64 threads (2 waves) per image.
//  - deterministic first-64 valid pixels (label==1, fill 0)
//  - gather sampled vectors -> LDS bf16
//  - sim = tf @ tf^T / TEMP via WMMA (2 waves x 8 tiles) -> LDS f32
//  - exp / log-prob / masked weighted sum -> atomicAdd scalar
// ---------------------------------------------------------------------------
__global__ __launch_bounds__(64) void contrastive_loss(
    const float* __restrict__ fbuf, const int* __restrict__ labels,
    const float* __restrict__ attnw, float* __restrict__ out) {
    int b = blockIdx.x, tid = threadIdx.x;
    __shared__ int sidx[NS];
    __shared__ float twl[NS];
    __shared__ __bf16 tfs[NS][PD];
    __shared__ float simm[NS][NS];
    __shared__ float red[NS];

    if (tid == 0) {
        int cnt = 0;
        for (int i = 0; i < HW && cnt < NS; ++i)
            if (labels[b * HW + i] == 1) sidx[cnt++] = i;
        for (; cnt < NS; ++cnt) sidx[cnt] = 0;
    }
    __syncthreads();

    int p = b * HW + sidx[tid];
    twl[tid] = attnw[p];
    for (int c = 0; c < PD; ++c) tfs[tid][c] = (__bf16)fbuf[(size_t)p * PD + c];
    __syncthreads();

    int wave = tid >> 5, lane = tid & 31;
    int hl = lane >> 4, m = lane & 15;
    v8f sacc[2][4];
#pragma unroll
    for (int i = 0; i < 2; ++i)
#pragma unroll
        for (int j = 0; j < 4; ++j)
            sacc[i][j] = (v8f){0.f, 0.f, 0.f, 0.f, 0.f, 0.f, 0.f, 0.f};

#pragma unroll
    for (int kb = 0; kb < 4; ++kb) {
        v16bf af[2], bf_[4];
#pragma unroll
        for (int tmi = 0; tmi < 2; ++tmi) {
            int row = 16 * (wave * 2 + tmi) + m;
#pragma unroll
            for (int j = 0; j < 16; ++j)
                af[tmi][j] = tfs[row][kb * 32 + a_kidx(j, hl)];
        }
#pragma unroll
        for (int tn = 0; tn < 4; ++tn) {
            int row = 16 * tn + m;
#pragma unroll
            for (int j = 0; j < 16; ++j)
                bf_[tn][j] = tfs[row][kb * 32 + b_kidx(j, hl)];
        }
#pragma unroll
        for (int tmi = 0; tmi < 2; ++tmi)
#pragma unroll
            for (int tn = 0; tn < 4; ++tn)
                sacc[tmi][tn] = __builtin_amdgcn_wmma_f32_16x16x32_bf16(
                    false, af[tmi], false, bf_[tn], (short)0, sacc[tmi][tn],
                    false, false);
    }

    const float invT = 1.f / TEMP;
#pragma unroll
    for (int tmi = 0; tmi < 2; ++tmi)
#pragma unroll
        for (int tn = 0; tn < 4; ++tn)
#pragma unroll
            for (int r = 0; r < 8; ++r) {
                int row = 16 * (wave * 2 + tmi) + r + 8 * hl;
                int cc = 16 * tn + m;
                simm[row][cc] = sacc[tmi][tn][r] * invT;
            }
    __syncthreads();

    // per-row loss (row = tid)
    float d = 0.f;
    for (int c = 0; c < NS; ++c) d += __expf(simm[tid][c]);
    float wi = twl[tid];
    float ls = 0.f;
    for (int c = 0; c < NS; ++c) {
        float s0 = simm[tid][c];
        if (s0 > SIM_TH && c != tid) {
            float lp = __logf(__expf(s0) / d + 1e-10f);
            ls += wi * twl[c] * lp;
        }
    }
    red[tid] = ls;
    __syncthreads();
    if (tid == 0) {
        float tot = 0.f, tws = 0.f;
        for (int i = 0; i < NS; ++i) { tot += red[i]; tws += twl[i]; }
        float loss = -(TEMP / BASE_TEMP) * tot / tws;
        atomicAdd(out, loss * (LOSS_W / (float)BATCH));
    }
}

__global__ void zero_out(float* o) {
    if (threadIdx.x == 0 && blockIdx.x == 0) o[0] = 0.f;
}

// ---------------------------------------------------------------------------
extern "C" void kernel_launch(void* const* d_in, const int* in_sizes, int n_in,
                              void* d_out, int out_size, void* d_ws, size_t ws_size,
                              hipStream_t stream) {
    const float* feats = (const float*)d_in[0];
    const int* labels = (const int*)d_in[1];
    const float* W1 = (const float*)d_in[2];
    const float* b1 = (const float*)d_in[3];
    const float* gamma = (const float*)d_in[4];
    const float* beta = (const float*)d_in[5];
    const float* W2 = (const float*)d_in[6];
    const float* b2 = (const float*)d_in[7];
    const float* Wa1 = (const float*)d_in[8];
    const float* ba1 = (const float*)d_in[9];
    const float* Wa2 = (const float*)d_in[10];
    const float* ba2 = (const float*)d_in[11];
    float* out = (float*)d_out;

    char* ws = (char*)d_ws;
    __bf16* wswz = (__bf16*)(ws + OFF_WSWZ);
    __bf16* w2swz = (__bf16*)(ws + OFF_W2SWZ);
    float* scale = (float*)(ws + OFF_SCALE);
    float* shift = (float*)(ws + OFF_SHIFT);
    float* attnw = (float*)(ws + OFF_ATTN);
    float* hbuf = (float*)(ws + OFF_H);
    float* fbuf = (float*)(ws + OFF_F);

    swizzle_weights<<<12 * 64 + 8 * 4, 32, 0, stream>>>(W1, Wa1, W2, wswz, w2swz);
    conv1_attn<<<NPIX / 16, 32, 0, stream>>>(feats, b1, ba1, Wa2, ba2, wswz,
                                             hbuf, attnw);
    bn_stats<<<PD, 256, 0, stream>>>(hbuf, gamma, beta, scale, shift);
    conv2_norm<<<NPIX / 16, 32, 0, stream>>>(hbuf, scale, shift, b2, w2swz, fbuf);
    zero_out<<<1, 1, 0, stream>>>(out);
    contrastive_loss<<<BATCH, 64, 0, stream>>>(fbuf, labels, attnw, out);
}
